// GatedDeltaProduct_45732811768030
// MI455X (gfx1250) — compile-verified, matrix operand order checked by reading, not run on hardware
//
#include <hip/hip_runtime.h>

typedef unsigned short u16;
typedef __bf16 v16bf __attribute__((ext_vector_type(16)));
typedef float v8f __attribute__((ext_vector_type(8)));
typedef unsigned int v8u __attribute__((ext_vector_type(8)));
typedef unsigned int tdm_v4u __attribute__((ext_vector_type(4)));
typedef int tdm_v4i __attribute__((ext_vector_type(4)));
typedef int tdm_v8i __attribute__((ext_vector_type(8)));

#if defined(__HIP_DEVICE_COMPILE__) && defined(__has_builtin)
#if __has_builtin(__builtin_amdgcn_tensor_load_to_lds)
#define HAVE_TDM 1
#endif
#endif
#ifndef HAVE_TDM
#define HAVE_TDM 0
#endif

// ---- problem constants ----
constexpr int cB   = 2;
constexpr int cT   = 512;
constexpr int cHID = 2048;
constexpr int cH   = 6;
constexpr int cDK  = 256;
constexpr int cNH  = 2;
constexpr int cKC  = 4;
constexpr int cDV  = 512;
constexpr int cKD  = cH * cDK;   // 1536
constexpr int cVD  = cH * cDV;   // 3072
constexpr int cM   = cB * cT;    // 1024

__device__ __forceinline__ u16 f32_bf16(float x) {
  unsigned u = __float_as_uint(x);
  u += 0x7FFFu + ((u >> 16) & 1u);
  return (u16)(u >> 16);
}

// ---------------------------------------------------------------------------
// f32 -> bf16 RNE converter (for the activation matrices)
// ---------------------------------------------------------------------------
__global__ void f32_to_bf16_kernel(const float* __restrict__ s, u16* __restrict__ d, size_t n) {
  size_t i = (size_t)blockIdx.x * blockDim.x + threadIdx.x;
  size_t stride = (size_t)gridDim.x * blockDim.x;
  for (; i < n; i += stride) d[i] = f32_bf16(s[i]);
}

// ---------------------------------------------------------------------------
// weight transpose + bf16 convert: w[K][N] f32 -> wt[N][K] bf16 (32x32 LDS tiles)
// ---------------------------------------------------------------------------
__global__ void transpose_cvt_kernel(const float* __restrict__ w, u16* __restrict__ wt,
                                     int K, int N) {
  __shared__ float tile[32][33];
  const int k0 = blockIdx.y * 32, n0 = blockIdx.x * 32;
  const int tx = threadIdx.x & 31, ty = threadIdx.x >> 5;  // 32 x 8
#pragma unroll
  for (int j = 0; j < 4; ++j)
    tile[ty + j * 8][tx] = w[(size_t)(k0 + ty + j * 8) * N + n0 + tx];
  __syncthreads();
#pragma unroll
  for (int j = 0; j < 4; ++j)
    wt[(size_t)(n0 + ty + j * 8) * K + k0 + tx] = f32_bf16(tile[tx][ty + j * 8]);
}

// ---------------------------------------------------------------------------
// TDM: DMA a rows x 64 bf16 tile (row stride K elems in memory) into LDS,
// padding each 128B row with 16B so LDS rows are 72 u16 (bank-conflict-free
// fragment loads). D# layout per CDNA5 ISA 8.3/8.4.
// ---------------------------------------------------------------------------
#if HAVE_TDM
__device__ __forceinline__ void tdm_load_tile(unsigned lds_off, const u16* gptr,
                                              int rows, int strideElems) {
  const unsigned long long ga = (unsigned long long)(uintptr_t)gptr;
  tdm_v4u g0;
  g0[0] = 1u;                                   // count=1, user descriptor
  g0[1] = lds_off;                              // lds_addr
  g0[2] = (unsigned)ga;                         // global_addr[31:0]
  g0[3] = (unsigned)((ga >> 32) & 0x1FFFFFFu) | (2u << 30);  // addr[56:32] | type=2
  const unsigned K32 = (unsigned)strideElems;
  const unsigned R32 = (unsigned)rows;
  v8u g1;
  g1[0] = (1u << 16)        // data_size = 2 bytes
        | (1u << 20)        // pad_enable
        | (4u << 22)        // pad_interval: 32 DWORDs (=128B row)
        | (3u << 25);       // pad_amount: 4 DWORDs (=16B)
  g1[1] = (K32 & 0xFFFFu) << 16;                       // tensor_dim0 lo16
  g1[2] = (K32 >> 16) | ((R32 & 0xFFFFu) << 16);       // dim0 hi16 | tensor_dim1 lo16
  g1[3] = (R32 >> 16) | (64u << 16);                   // dim1 hi16 | tile_dim0=64
  g1[4] = R32;                                         // tile_dim1=rows, tile_dim2=0
  g1[5] = K32;                                         // tensor_dim0_stride lo32
  g1[6] = 0u;
  g1[7] = 0u;
  const tdm_v4i z4 = {0, 0, 0, 0};
#if __clang_major__ >= 23
  const tdm_v8i z8 = {0, 0, 0, 0, 0, 0, 0, 0};
  __builtin_amdgcn_tensor_load_to_lds(g0, __builtin_bit_cast(tdm_v8i, g1), z4, z4, z8, 0);
#else
  __builtin_amdgcn_tensor_load_to_lds(g0, __builtin_bit_cast(tdm_v8i, g1), z4, z4, 0);
#endif
}
#endif

// ---------------------------------------------------------------------------
// bf16 WMMA GEMM, C[cM,N] = A[cM,K] * Bt[N,K]^T.  Compile-time N,K.
// Block: 256 threads / 8 waves; block tile 64x128; wave tile 32x32 (2x2 frags);
// BK=64 (two WMMA k-steps per stage). Both tiles land in LDS with 72-u16 rows.
// TDM path: double-buffered tensor_load_to_lds issued by wave 0, overlapped
// with WMMA via s_wait_tensorcnt(2).
// ---------------------------------------------------------------------------
constexpr int LDW = 72;

__device__ __forceinline__ v16bf load_frag(const u16 (*tile)[LDW], int r0, int kk, int lane) {
  const int m = lane & 15, hi = lane >> 4;
  v8u u;
#pragma unroll
  for (int j = 0; j < 8; ++j) {
    const int kp = ((j < 4) ? (2 * j) : (16 + 2 * (j - 4))) + 8 * hi + kk;
    u[j] = *(const unsigned int*)&tile[r0 + m][kp];
  }
  return __builtin_bit_cast(v16bf, u);
}

template <int N, int K>
__global__ void __launch_bounds__(256, 2) wmma_gemm_bf16(
    const u16* __restrict__ A, const u16* __restrict__ Btg, float* __restrict__ C) {
  __shared__ u16 As[2][64][LDW];
  __shared__ u16 Bs[2][128][LDW];

  const int tid  = threadIdx.x;
  const int lane = tid & 31, wid = tid >> 5;
  const int waveM = wid >> 2, waveN = wid & 3;
  const int rowBase = blockIdx.y * 64;
  const int colBase = blockIdx.x * 128;

  v8f acc[2][2];
#pragma unroll
  for (int i = 0; i < 2; ++i)
#pragma unroll
    for (int j = 0; j < 2; ++j)
#pragma unroll
      for (int r = 0; r < 8; ++r) acc[i][j][r] = 0.f;

#if HAVE_TDM
  if (wid == 0) {
    tdm_load_tile((unsigned)(uintptr_t)&As[0][0][0], A + (size_t)rowBase * K, 64, K);
    tdm_load_tile((unsigned)(uintptr_t)&Bs[0][0][0], Btg + (size_t)colBase * K, 128, K);
  }
  int cur = 0;
  for (int k0 = 0; k0 < K; k0 += 64) {
    const int nxt = cur ^ 1;
    if (wid == 0) {
      if (k0 + 64 < K) {
        tdm_load_tile((unsigned)(uintptr_t)&As[nxt][0][0],
                      A + (size_t)rowBase * K + (k0 + 64), 64, K);
        tdm_load_tile((unsigned)(uintptr_t)&Bs[nxt][0][0],
                      Btg + (size_t)colBase * K + (k0 + 64), 128, K);
        __builtin_amdgcn_s_wait_tensorcnt(2);  // wait for cur pair, nxt in flight
      } else {
        __builtin_amdgcn_s_wait_tensorcnt(0);
      }
    }
    __syncthreads();
#pragma unroll
    for (int kk = 0; kk < 64; kk += 32) {
      v16bf af[2], bfr[2];
#pragma unroll
      for (int i = 0; i < 2; ++i) af[i] = load_frag(As[cur], waveM * 32 + i * 16, kk, lane);
#pragma unroll
      for (int j = 0; j < 2; ++j) bfr[j] = load_frag(Bs[cur], waveN * 32 + j * 16, kk, lane);
#pragma unroll
      for (int i = 0; i < 2; ++i)
#pragma unroll
        for (int j = 0; j < 2; ++j)
          acc[i][j] = __builtin_amdgcn_wmma_f32_16x16x32_bf16(
              false, af[i], false, bfr[j], (short)0, acc[i][j], false, false);
    }
    __syncthreads();
    cur = nxt;
  }
#else
  // manual fallback: vectorized loads, both tiles row-major (B pre-transposed)
  const int ar = tid >> 2, ac = (tid & 3) * 16;  // A: 64 rows x 64 cols
  for (int k0 = 0; k0 < K; k0 += 64) {
    *(uint4*)&As[0][ar][ac]     = *(const uint4*)(A + (size_t)(rowBase + ar) * K + k0 + ac);
    *(uint4*)&As[0][ar][ac + 8] = *(const uint4*)(A + (size_t)(rowBase + ar) * K + k0 + ac + 8);
#pragma unroll
    for (int rep = 0; rep < 2; ++rep) {
      const int idx = tid + rep * 256;
      const int nr = idx >> 2, kc = (idx & 3) * 16;
      *(uint4*)&Bs[0][nr][kc]     = *(const uint4*)(Btg + (size_t)(colBase + nr) * K + k0 + kc);
      *(uint4*)&Bs[0][nr][kc + 8] = *(const uint4*)(Btg + (size_t)(colBase + nr) * K + k0 + kc + 8);
    }
    __syncthreads();
#pragma unroll
    for (int kk = 0; kk < 64; kk += 32) {
      v16bf af[2], bfr[2];
#pragma unroll
      for (int i = 0; i < 2; ++i) af[i] = load_frag(As[0], waveM * 32 + i * 16, kk, lane);
#pragma unroll
      for (int j = 0; j < 2; ++j) bfr[j] = load_frag(Bs[0], waveN * 32 + j * 16, kk, lane);
#pragma unroll
      for (int i = 0; i < 2; ++i)
#pragma unroll
        for (int j = 0; j < 2; ++j)
          acc[i][j] = __builtin_amdgcn_wmma_f32_16x16x32_bf16(
              false, af[i], false, bfr[j], (short)0, acc[i][j], false, false);
    }
    __syncthreads();
  }
#endif

  // C/D layout: lanes 0-15 -> M = r, lanes 16-31 -> M = 8+r; N = lane&15
  const int n = lane & 15, mb = (lane >> 4) * 8;
#pragma unroll
  for (int i = 0; i < 2; ++i)
#pragma unroll
    for (int j = 0; j < 2; ++j) {
      float* cp = C + (size_t)(rowBase + waveM * 32 + i * 16 + mb) * N +
                  (colBase + waveN * 32 + j * 16 + n);
#pragma unroll
      for (int r = 0; r < 8; ++r) cp[r * N] = acc[i][j][r];  // immediate offsets
    }
}

// ---------------------------------------------------------------------------
// small projections: beta = sigmoid(h @ Wb), gdecay = -exp(A_log)*softplus(h@Wa+dt_bias)
// ---------------------------------------------------------------------------
__global__ void smallproj_kernel(const float* __restrict__ hid, const float* __restrict__ Wb,
                                 const float* __restrict__ Wa, const float* __restrict__ A_log,
                                 const float* __restrict__ dtb, float* __restrict__ beta,
                                 float* __restrict__ gdec) {
  const int gid = blockIdx.x * blockDim.x + threadIdx.x;
  if (gid >= cM * (cNH * cH + cH)) return;
  const int row = gid / (cNH * cH + cH);
  const int c = gid % (cNH * cH + cH);
  const float* hrow = hid + (size_t)row * cHID;
  if (c < cNH * cH) {
    const int head = c / cH, h = c % cH;
    const float* w = Wb + (size_t)head * cHID * cH + h;
    float acc = 0.f;
    for (int k = 0; k < cHID; ++k) acc += hrow[k] * w[(size_t)k * cH];
    beta[(size_t)head * cM * cH + (size_t)row * cH + h] = 1.f / (1.f + __expf(-acc));
  } else {
    const int h = c - cNH * cH;
    float acc = 0.f;
    for (int k = 0; k < cHID; ++k) acc += hrow[k] * Wa[(size_t)k * cH + h];
    const float x = acc + dtb[h];
    const float sp = (x > 20.f) ? x : log1pf(__expf(x));
    gdec[(size_t)row * cH + h] = -__expf(A_log[h]) * sp;
  }
}

// ---------------------------------------------------------------------------
// causal depthwise conv1d (K=4) + SiLU. x,y: [B,T,C], w: [C,4]
// ---------------------------------------------------------------------------
__global__ void conv_silu_kernel(const float* __restrict__ x, const float* __restrict__ w,
                                 float* __restrict__ y, int C, size_t total) {
  size_t i = (size_t)blockIdx.x * blockDim.x + threadIdx.x;
  const size_t stride = (size_t)gridDim.x * blockDim.x;
  for (; i < total; i += stride) {
    const int c = (int)(i % C);
    const size_t bt = i / C;
    const int t = (int)(bt % cT);
    const size_t brow = bt - t;
    float acc = 0.f;
#pragma unroll
    for (int j = 0; j < cKC; ++j) {
      const int ts = t - (cKC - 1) + j;
      const float xv = (ts >= 0) ? x[(brow + ts) * C + c] : 0.f;
      acc += w[(size_t)c * cKC + j] * xv;
    }
    y[i] = acc / (1.f + __expf(-acc));
  }
}

// ---------------------------------------------------------------------------
// per-(row, head) L2 norm over DK=256, in place
// ---------------------------------------------------------------------------
__global__ void l2norm_kernel(float* __restrict__ x, float scale) {
  const int row = blockIdx.x, h = blockIdx.y;
  float* p = x + (size_t)row * cKD + h * cDK;
  const int tid = threadIdx.x;
  const float v = p[tid];
  float s = v * v;
#pragma unroll
  for (int off = 16; off > 0; off >>= 1) s += __shfl_down(s, off, 32);
  __shared__ float red[8];
  if ((tid & 31) == 0) red[tid >> 5] = s;
  __syncthreads();
  __shared__ float rsv;
  if (tid == 0) {
    float tot = 0.f;
#pragma unroll
    for (int i = 0; i < 8; ++i) tot += red[i];
    rsv = rsqrtf(tot + 1e-6f) * scale;
  }
  __syncthreads();
  p[tid] = v * rsv;
}

// ---------------------------------------------------------------------------
// gated delta-rule scan. 512 threads: (vc = tid&255) x (khalf = tid>>8).
// Each thread holds a 128-deep half-column of S in VGPRs (no spill);
// partial dots/outputs combine via LDS. Grid: (DV/256, H, B).
// ---------------------------------------------------------------------------
__global__ void __launch_bounds__(512, 1) scan_kernel(
    const float* __restrict__ qn, const float* __restrict__ kn0, const float* __restrict__ kn1,
    const float* __restrict__ v0p, const float* __restrict__ v1p,
    const float* __restrict__ beta0, const float* __restrict__ beta1,
    const float* __restrict__ gdec, float* __restrict__ o) {
  const int tid = threadIdx.x;
  const int vc = tid & 255;
  const int kh = tid >> 8;
  const int vcol = blockIdx.x * 256 + vc;
  const int h = blockIdx.y, b = blockIdx.z;
  const int kb = kh * 128;
  __shared__ float sk[cDK];
  __shared__ float sq[cDK];
  __shared__ float part[2][256];

  float S[128];
#pragma unroll
  for (int k = 0; k < 128; ++k) S[k] = 0.f;

  for (int t = 0; t < cT; ++t) {
    const int row = b * cT + t;
    const size_t kbase = (size_t)row * cKD + h * cDK;
    // ---- phase 0 : head 0, decay active ----
    if (tid < cDK) sk[tid] = kn0[kbase + tid];
    __syncthreads();
    const float eg  = __expf(gdec[(size_t)row * cH + h]);
    const float b0  = beta0[(size_t)row * cH + h];
    const float vt0 = v0p[(size_t)row * cVD + h * cDV + vcol];
    float acc = 0.f;
#pragma unroll
    for (int k = 0; k < 128; ++k) acc += sk[kb + k] * S[k];
    part[kh][vc] = acc;
    __syncthreads();
    const float d0 = b0 * (vt0 - eg * (part[0][vc] + part[1][vc]));
#pragma unroll
    for (int k = 0; k < 128; ++k) S[k] = eg * S[k] + sk[kb + k] * d0;
    __syncthreads();
    // ---- phase 1 : head 1, output with q ----
    if (tid < cDK) { sk[tid] = kn1[kbase + tid]; sq[tid] = qn[kbase + tid]; }
    __syncthreads();
    const float b1  = beta1[(size_t)row * cH + h];
    const float vt1 = v1p[(size_t)row * cVD + h * cDV + vcol];
    acc = 0.f;
#pragma unroll
    for (int k = 0; k < 128; ++k) acc += sk[kb + k] * S[k];
    part[kh][vc] = acc;
    __syncthreads();
    const float d1 = b1 * (vt1 - (part[0][vc] + part[1][vc]));
    __syncthreads();  // all reads of part done before reuse
    float oa = 0.f;
#pragma unroll
    for (int k = 0; k < 128; ++k) {
      S[k] += sk[kb + k] * d1;
      oa += sq[kb + k] * S[k];
    }
    part[kh][vc] = oa;
    __syncthreads();
    if (kh == 0) o[((size_t)row * cH + h) * cDV + vcol] = part[0][vc] + part[1][vc];
    __syncthreads();
  }
}

// ---------------------------------------------------------------------------
// gated RMSNorm over DV * o_norm_w, swish-gated; emit bf16 for final GEMM
// ---------------------------------------------------------------------------
__global__ void postgate_kernel(const float* __restrict__ o, const float* __restrict__ g,
                                const float* __restrict__ onw, u16* __restrict__ out) {
  const int idx = blockIdx.x;  // row*H + h
  const int row = idx / cH, h = idx % cH;
  const int tid = threadIdx.x;
  const float* op = o + (size_t)idx * cDV;
  const float a0 = op[tid], a1 = op[tid + 256];
  float s = a0 * a0 + a1 * a1;
#pragma unroll
  for (int off = 16; off > 0; off >>= 1) s += __shfl_down(s, off, 32);
  __shared__ float red[8];
  if ((tid & 31) == 0) red[tid >> 5] = s;
  __syncthreads();
  __shared__ float rsv;
  if (tid == 0) {
    float tot = 0.f;
#pragma unroll
    for (int i = 0; i < 8; ++i) tot += red[i];
    rsv = rsqrtf(tot / (float)cDV + 1e-5f);
  }
  __syncthreads();
  const float rs = rsv;
  const size_t gbase = (size_t)row * cVD + h * cDV;
#pragma unroll
  for (int e = 0; e < 2; ++e) {
    const int v = tid + e * 256;
    const float a = e ? a1 : a0;
    const float nv = a * rs * onw[v];
    const float gg = g[gbase + v];
    out[gbase + v] = f32_bf16(nv * gg / (1.f + __expf(-gg)));
  }
}

// ---------------------------------------------------------------------------
// launcher
// ---------------------------------------------------------------------------
extern "C" void kernel_launch(void* const* d_in, const int* in_sizes, int n_in,
                              void* d_out, int out_size, void* d_ws, size_t ws_size,
                              hipStream_t stream) {
  const float* hid   = (const float*)d_in[0];
  const float* Wq    = (const float*)d_in[1];
  const float* Wk    = (const float*)d_in[2];
  const float* Wv    = (const float*)d_in[3];
  const float* Wb    = (const float*)d_in[4];
  const float* Wa    = (const float*)d_in[5];
  const float* Wg    = (const float*)d_in[6];
  const float* Wo    = (const float*)d_in[7];
  const float* cqw   = (const float*)d_in[8];
  const float* ckw   = (const float*)d_in[9];
  const float* cvw   = (const float*)d_in[10];
  const float* A_log = (const float*)d_in[11];
  const float* dtb   = (const float*)d_in[12];
  const float* onw   = (const float*)d_in[13];

  char* p = (char*)d_ws;
  auto alloc = [&](size_t bytes) -> char* {
    char* r = p;
    p += (bytes + 255) & ~(size_t)255;
    return r;
  };

  u16*   hid_bf  = (u16*)alloc((size_t)cM * cHID * 2);
  u16*   Wq_t    = (u16*)alloc((size_t)cKD * cHID * 2);
  u16*   Wk_t    = (u16*)alloc((size_t)cNH * cKD * cHID * 2);
  u16*   Wv_t    = (u16*)alloc((size_t)cNH * cVD * cHID * 2);
  u16*   Wg_t    = (u16*)alloc((size_t)cVD * cHID * 2);
  u16*   Wo_t    = (u16*)alloc((size_t)cHID * cVD * 2);
  float* qproj   = (float*)alloc((size_t)cM * cKD * 4);
  float* kproj   = (float*)alloc((size_t)cNH * cM * cKD * 4);
  float* vproj   = (float*)alloc((size_t)cNH * cM * cVD * 4);
  float* gproj   = (float*)alloc((size_t)cM * cVD * 4);
  float* qcv     = (float*)alloc((size_t)cM * cKD * 4);
  float* kcv     = (float*)alloc((size_t)cNH * cM * cKD * 4);
  float* vcv     = (float*)alloc((size_t)cNH * cM * cVD * 4);
  float* betab   = (float*)alloc((size_t)cNH * cM * cH * 4);
  float* gdec    = (float*)alloc((size_t)cM * cH * 4);
  float* obuf    = (float*)alloc((size_t)cM * cH * cDV * 4);
  u16*   gate_bf = (u16*)alloc((size_t)cM * cVD * 2);

  // activations -> bf16
  f32_to_bf16_kernel<<<2048, 256, 0, stream>>>(hid, hid_bf, (size_t)cM * cHID);

  // weights -> transposed bf16 [N][K]
  auto tcvt = [&](const float* w, u16* wt, int K, int N) {
    transpose_cvt_kernel<<<dim3(N / 32, K / 32), 256, 0, stream>>>(w, wt, K, N);
  };
  tcvt(Wq, Wq_t, cHID, cKD);
  for (int i = 0; i < cNH; ++i)
    tcvt(Wk + (size_t)i * cHID * cKD, Wk_t + (size_t)i * cKD * cHID, cHID, cKD);
  for (int i = 0; i < cNH; ++i)
    tcvt(Wv + (size_t)i * cHID * cVD, Wv_t + (size_t)i * cVD * cHID, cHID, cVD);
  tcvt(Wg, Wg_t, cHID, cVD);
  tcvt(Wo, Wo_t, cVD, cHID);

  // projection GEMMs
  wmma_gemm_bf16<cKD, cHID><<<dim3(cKD / 128, cM / 64), 256, 0, stream>>>(hid_bf, Wq_t, qproj);
  for (int i = 0; i < cNH; ++i)
    wmma_gemm_bf16<cKD, cHID><<<dim3(cKD / 128, cM / 64), 256, 0, stream>>>(
        hid_bf, Wk_t + (size_t)i * cKD * cHID, kproj + (size_t)i * cM * cKD);
  for (int i = 0; i < cNH; ++i)
    wmma_gemm_bf16<cVD, cHID><<<dim3(cVD / 128, cM / 64), 256, 0, stream>>>(
        hid_bf, Wv_t + (size_t)i * cVD * cHID, vproj + (size_t)i * cM * cVD);
  wmma_gemm_bf16<cVD, cHID><<<dim3(cVD / 128, cM / 64), 256, 0, stream>>>(hid_bf, Wg_t, gproj);

  // beta / gdecay
  smallproj_kernel<<<(cM * (cNH * cH + cH) + 255) / 256, 256, 0, stream>>>(
      hid, Wb, Wa, A_log, dtb, betab, gdec);

  // causal convs + SiLU
  {
    const size_t nq = (size_t)cM * cKD;
    conv_silu_kernel<<<2048, 256, 0, stream>>>(qproj, cqw, qcv, cKD, nq);
    for (int i = 0; i < cNH; ++i)
      conv_silu_kernel<<<2048, 256, 0, stream>>>(kproj + (size_t)i * cM * cKD,
                                                 ckw + (size_t)i * cKD * cKC,
                                                 kcv + (size_t)i * cM * cKD, cKD, nq);
    const size_t nv = (size_t)cM * cVD;
    for (int i = 0; i < cNH; ++i)
      conv_silu_kernel<<<4096, 256, 0, stream>>>(vproj + (size_t)i * cM * cVD,
                                                 cvw + (size_t)i * cVD * cKC,
                                                 vcv + (size_t)i * cM * cVD, cVD, nv);
  }

  // L2 norms (q scaled by DK^-0.5 = 1/16)
  l2norm_kernel<<<dim3(cM, cH), 256, 0, stream>>>(qcv, 0.0625f);
  l2norm_kernel<<<dim3(cM, cH), 256, 0, stream>>>(kcv, 1.0f);
  l2norm_kernel<<<dim3(cM, cH), 256, 0, stream>>>(kcv + (size_t)cM * cKD, 1.0f);

  // gated delta-rule scan
  scan_kernel<<<dim3(cDV / 256, cH, cB), 512, 0, stream>>>(
      qcv, kcv, kcv + (size_t)cM * cKD, vcv, vcv + (size_t)cM * cVD,
      betab, betab + (size_t)cM * cH, gdec, obuf);

  // gated RMSNorm -> bf16
  postgate_kernel<<<cM * cH, 256, 0, stream>>>(obuf, gproj, onw, gate_bf);

  // output projection
  wmma_gemm_bf16<cHID, cVD><<<dim3(cHID / 128, cM / 64), 256, 0, stream>>>(
      gate_bf, Wo_t, (float*)d_out);
}
